// NVFP4MoEModuleForInputScaleTest_20839181320307
// MI455X (gfx1250) — matrix operand order repaired and checked by reading, not compiled, not run on hardware
//
#include <hip/hip_runtime.h>

// ---------------------------------------------------------------------------
// MoE gated MLP (NVFP4-sim) for MI455X / gfx1250, wave32 + WMMA f16.
//   T=8192 tokens, H=I=2048, E=8 experts, top-K=2.
// Routed (gathered) grouped GEMMs in f16 with f32 WMMA accumulation.
// Global->LDS staging uses CDNA5 async-to-LDS DMA (ASYNCcnt) with explicit
// double buffering: DMA panel k+1 while WMMAs consume panel k.
// ---------------------------------------------------------------------------

#define TT 8192
#define HH 2048
#define II 2048
#define EE 8
#define KK 2
#define TK (TT * KK)          // 16384 gathered rows total

#define BM 128                // block tile rows (gathered tokens)
#define BN 128                // block tile cols
#define BK 32                 // K panel depth (matches 16x16x32 WMMA)
#define LDST 40               // padded LDS k-stride in halves (80B, 16B aligned)
#define MAXRT (TK / BM)       // worst-case row tiles per expert = 128

typedef __attribute__((ext_vector_type(16))) _Float16 v16h;
typedef __attribute__((ext_vector_type(8)))  float    v8f;

union FragU { uint4 q[2]; v16h v; };

static __device__ __forceinline__ int imin(int a, int b) { return a < b ? a : b; }

// --- CDNA5 async global->LDS DMA (16B per lane), tracked by ASYNCcnt -------
// lds_addr: 32-bit LDS byte offset (generic shared ptr truncated: flat->LDS
// mapping keeps the byte offset in addr[31:0]); gaddr: 64-bit global address.
static __device__ __forceinline__ void async_ld16(unsigned lds_addr, const void* gsrc) {
    unsigned long long ga = (unsigned long long)gsrc;
    asm volatile("global_load_async_to_lds_b128 %0, %1, off"
                 :: "v"(lds_addr), "v"(ga) : "memory");
}
#define S_WAIT_ASYNC(imm_str) asm volatile("s_wait_asynccnt " imm_str ::: "memory")

static __device__ __forceinline__ unsigned lds_off(const void* p) {
    return (unsigned)(unsigned long long)p;
}

// A-matrix 16x32 (f16) fragment per ISA layout:
//   lanes 0-15 : row M=lane,    halves = K[0..7]  ++ K[16..23]
//   lanes16-31 : row M=lane-16, halves = K[8..15] ++ K[24..31]
static __device__ __forceinline__ v16h ld_frag_a(const _Float16* lds, int row, int lane) {
    int koff = (lane >= 16) ? 8 : 0;
    FragU f;
    f.q[0] = *(const uint4*)(lds + row * LDST + koff);
    f.q[1] = *(const uint4*)(lds + row * LDST + koff + 16);
    return f.v;
}

// B-matrix 32x16 (f16) fragment: lane n<16 holds column n, K[0..7] ++ K[16..23];
// lane n>=16 holds column n-16, K[8..15] ++ K[24..31]. B[k][n] = w[n][k].
static __device__ __forceinline__ v16h ld_frag_b(const _Float16* lds, int col, int lane) {
    int koff = (lane >= 16) ? 8 : 0;
    FragU f;
    f.q[0] = *(const uint4*)(lds + col * LDST + koff);
    f.q[1] = *(const uint4*)(lds + col * LDST + koff + 16);
    return f.v;
}

static __device__ __forceinline__ v8f wmma16(v16h a, v16h b, v8f c) {
    return __builtin_amdgcn_wmma_f32_16x16x32_f16(
        /*neg_a=*/false, a, /*neg_b=*/false, b,
        /*c_mod=*/(short)0, c, /*reuse_a=*/false, /*reuse_b=*/false);
}

// ------------------------------- routing -----------------------------------

__global__ void moe_zero8(int* counts) {
    if (threadIdx.x < EE) counts[threadIdx.x] = 0;
}

__global__ void moe_count(const int* __restrict__ sel, int* __restrict__ counts) {
    int i = blockIdx.x * blockDim.x + threadIdx.x;
    if (i < TK) atomicAdd(&counts[sel[i]], 1);
}

__global__ void moe_scan(const int* __restrict__ counts,
                         int* __restrict__ offsets, int* __restrict__ cursors) {
    if (blockIdx.x == 0 && threadIdx.x == 0) {
        int acc = 0;
        for (int e = 0; e < EE; ++e) {
            offsets[e] = acc; cursors[e] = acc; acc += counts[e];
        }
        offsets[EE] = acc;
    }
}

__global__ void moe_fill(const int* __restrict__ sel,
                         int* __restrict__ cursors,
                         int* __restrict__ rows, int* __restrict__ inv) {
    int i = blockIdx.x * blockDim.x + threadIdx.x;
    if (i < TK) {
        int e = sel[i];
        int p = atomicAdd(&cursors[e], 1);
        rows[p] = i / KK;   // token id
        inv[i]  = p;        // where this (token,k) slot landed
    }
}

// --------------------------- f32 -> f16 convert ----------------------------

__global__ void moe_cvt(const float* __restrict__ s, _Float16* __restrict__ d, long long n4) {
    long long i  = (long long)blockIdx.x * blockDim.x + threadIdx.x;
    long long st = (long long)gridDim.x * blockDim.x;
    for (; i < n4; i += st) {
        float4 v = ((const float4*)s)[i];
        union { _Float16 h[4]; uint2 u; } o;
        o.h[0] = (_Float16)v.x; o.h[1] = (_Float16)v.y;
        o.h[2] = (_Float16)v.z; o.h[3] = (_Float16)v.w;
        ((uint2*)d)[i] = o.u;
    }
}

// ------------------------------- stage 1 -----------------------------------
// h[p, :] = silu(s1 * x_tok @ w1_e^T) * (s3 * x_tok @ w3_e^T), f16 out.

__global__ __launch_bounds__(256)
void moe_stage1(const _Float16* __restrict__ xb,
                const _Float16* __restrict__ w1b,
                const _Float16* __restrict__ w3b,
                const int* __restrict__ counts,
                const int* __restrict__ offsets,
                const int* __restrict__ rows,
                const float* __restrict__ w1_is, const float* __restrict__ w1_ws,
                const float* __restrict__ w1_al,
                const float* __restrict__ w3_is, const float* __restrict__ w3_ws,
                const float* __restrict__ w3_al,
                _Float16* __restrict__ hbuf) {
    const int e    = blockIdx.z;
    const int cnt  = counts[e];
    const int row0 = blockIdx.y * BM;
    if (row0 >= cnt) return;                 // uniform across block
    const int col0 = blockIdx.x * BN;
    const int base = offsets[e];

    // double-buffered panels
    __shared__ _Float16 lA [2 * BM * LDST];
    __shared__ _Float16 lB1[2 * BN * LDST];
    __shared__ _Float16 lB3[2 * BN * LDST];

    const int tid  = threadIdx.x;
    const int lane = tid & 31;
    const int wave = tid >> 5;
    const int wm   = (wave & 3) * 32;        // 4 waves down M (32 rows each)
    const int wn   = (wave >> 2) * 64;       // 2 waves across N (64 cols each)

    v8f accg[2][4], accu[2][4];
    #pragma unroll
    for (int mi = 0; mi < 2; ++mi)
        #pragma unroll
        for (int j = 0; j < 4; ++j) { accg[mi][j] = (v8f)0.0f; accu[mi][j] = (v8f)0.0f; }

    // per-thread cooperative DMA assignment: one 32B chunk per array
    const int lr  = tid >> 1;                // 0..127 : tile row
    const int seg = tid & 1;                 // 0..1   : 16-half segment
    const int gi  = imin(base + row0 + lr, TK - 1);
    const int tok = rows[gi];
    const size_t aRow  = (size_t)tok * HH + seg * 16;
    const size_t bRow  = ((size_t)e * II + col0 + lr) * HH + seg * 16;
    const unsigned dA0 = lds_off(lA)  + (unsigned)(lr * LDST + seg * 16) * 2u;
    const unsigned dB0 = lds_off(lB1) + (unsigned)(lr * LDST + seg * 16) * 2u;
    const unsigned dC0 = lds_off(lB3) + (unsigned)(lr * LDST + seg * 16) * 2u;
    const unsigned bufB = (unsigned)(BM * LDST) * 2u;    // bytes per buffer

    // issue panel kt's 6 async 16B DMAs into buffer b
    auto issue = [&](int kt, int b) {
        unsigned off = (unsigned)b * bufB;
        async_ld16(dA0 + off,      xb  + aRow + kt);
        async_ld16(dA0 + off + 16, xb  + aRow + kt + 8);
        async_ld16(dB0 + off,      w1b + bRow + kt);
        async_ld16(dB0 + off + 16, w1b + bRow + kt + 8);
        async_ld16(dC0 + off,      w3b + bRow + kt);
        async_ld16(dC0 + off + 16, w3b + bRow + kt + 8);
    };

    issue(0, 0);
    const int NK = HH / BK;
    for (int it = 0; it < NK; ++it) {
        if (it + 1 < NK) {
            issue((it + 1) * BK, (it + 1) & 1);
            S_WAIT_ASYNC("0x6");             // older panel done, next in flight
        } else {
            S_WAIT_ASYNC("0x0");
        }
        __syncthreads();                      // publish panel `it` to all waves

        const _Float16* cA  = lA  + (it & 1) * (BM * LDST);
        const _Float16* cB1 = lB1 + (it & 1) * (BN * LDST);
        const _Float16* cB3 = lB3 + (it & 1) * (BN * LDST);

        v16h a0 = ld_frag_a(cA, wm +      (lane & 15), lane);
        v16h a1 = ld_frag_a(cA, wm + 16 + (lane & 15), lane);
        #pragma unroll
        for (int j = 0; j < 4; ++j) {
            v16h b1 = ld_frag_b(cB1, wn + 16 * j + (lane & 15), lane);
            v16h b3 = ld_frag_b(cB3, wn + 16 * j + (lane & 15), lane);
            accg[0][j] = wmma16(a0, b1, accg[0][j]);
            accg[1][j] = wmma16(a1, b1, accg[1][j]);
            accu[0][j] = wmma16(a0, b3, accu[0][j]);
            accu[1][j] = wmma16(a1, b3, accu[1][j]);
        }
        __syncthreads();                      // protect buffer for next DMA
    }

    // out = alpha*((x/is)@(w*ws)^T) = (alpha*ws/is) * x@w^T
    const float s1 = w1_al[e] * w1_ws[e] / w1_is[e];
    const float s3 = w3_al[e] * w3_ws[e] / w3_is[e];

    #pragma unroll
    for (int mi = 0; mi < 2; ++mi) {
        #pragma unroll
        for (int j = 0; j < 4; ++j) {
            const int colg = col0 + wn + 16 * j + (lane & 15);
            #pragma unroll
            for (int r = 0; r < 8; ++r) {
                int rl   = wm + 16 * mi + ((lane < 16) ? r : r + 8);
                int grow = row0 + rl;
                if (grow < cnt) {
                    float g   = s1 * accg[mi][j][r];
                    float u   = s3 * accu[mi][j][r];
                    float sil = g / (1.0f + __expf(-g));
                    hbuf[(size_t)(base + grow) * II + colg] = (_Float16)(sil * u);
                }
            }
        }
    }
}

// ------------------------------- stage 2 -----------------------------------
// yb[p, :] = s2 * (h[p, :] @ w2_e^T), f32 out per gathered row.

__global__ __launch_bounds__(256)
void moe_stage2(const _Float16* __restrict__ hbuf,
                const _Float16* __restrict__ w2b,
                const int* __restrict__ counts,
                const int* __restrict__ offsets,
                const float* __restrict__ w2_is, const float* __restrict__ w2_ws,
                const float* __restrict__ w2_al,
                float* __restrict__ yb) {
    const int e    = blockIdx.z;
    const int cnt  = counts[e];
    const int row0 = blockIdx.y * BM;
    if (row0 >= cnt) return;
    const int col0 = blockIdx.x * BN;
    const int base = offsets[e];

    __shared__ _Float16 lA[2 * BM * LDST];
    __shared__ _Float16 lB[2 * BN * LDST];

    const int tid  = threadIdx.x;
    const int lane = tid & 31;
    const int wave = tid >> 5;
    const int wm   = (wave & 3) * 32;
    const int wn   = (wave >> 2) * 64;

    v8f acc[2][4];
    #pragma unroll
    for (int mi = 0; mi < 2; ++mi)
        #pragma unroll
        for (int j = 0; j < 4; ++j) acc[mi][j] = (v8f)0.0f;

    const int lr  = tid >> 1;
    const int seg = tid & 1;
    const size_t aRow = (size_t)imin(base + row0 + lr, TK - 1) * II + seg * 16;
    const size_t bRow = ((size_t)e * HH + col0 + lr) * II + seg * 16;
    const unsigned dA0 = lds_off(lA) + (unsigned)(lr * LDST + seg * 16) * 2u;
    const unsigned dB0 = lds_off(lB) + (unsigned)(lr * LDST + seg * 16) * 2u;
    const unsigned bufB = (unsigned)(BM * LDST) * 2u;

    auto issue = [&](int kt, int b) {
        unsigned off = (unsigned)b * bufB;
        async_ld16(dA0 + off,      hbuf + aRow + kt);
        async_ld16(dA0 + off + 16, hbuf + aRow + kt + 8);
        async_ld16(dB0 + off,      w2b  + bRow + kt);
        async_ld16(dB0 + off + 16, w2b  + bRow + kt + 8);
    };

    issue(0, 0);
    const int NK = II / BK;
    for (int it = 0; it < NK; ++it) {
        if (it + 1 < NK) {
            issue((it + 1) * BK, (it + 1) & 1);
            S_WAIT_ASYNC("0x4");
        } else {
            S_WAIT_ASYNC("0x0");
        }
        __syncthreads();

        const _Float16* cA = lA + (it & 1) * (BM * LDST);
        const _Float16* cB = lB + (it & 1) * (BN * LDST);

        v16h a0 = ld_frag_a(cA, wm +      (lane & 15), lane);
        v16h a1 = ld_frag_a(cA, wm + 16 + (lane & 15), lane);
        #pragma unroll
        for (int j = 0; j < 4; ++j) {
            v16h b = ld_frag_b(cB, wn + 16 * j + (lane & 15), lane);
            acc[0][j] = wmma16(a0, b, acc[0][j]);
            acc[1][j] = wmma16(a1, b, acc[1][j]);
        }
        __syncthreads();
    }

    const float s2 = w2_al[e] * w2_ws[e] / w2_is[e];

    #pragma unroll
    for (int mi = 0; mi < 2; ++mi) {
        #pragma unroll
        for (int j = 0; j < 4; ++j) {
            const int colg = col0 + wn + 16 * j + (lane & 15);
            #pragma unroll
            for (int r = 0; r < 8; ++r) {
                int rl   = wm + 16 * mi + ((lane < 16) ? r : r + 8);
                int grow = row0 + rl;
                if (grow < cnt)
                    yb[(size_t)(base + grow) * HH + colg] = s2 * acc[mi][j][r];
            }
        }
    }
}

// ------------------------------- combine -----------------------------------
// out[t, :] = sum_k rw[t,k] * yb[inv[t*K+k], :]   (deterministic, no atomics)

__global__ void moe_combine(const float* __restrict__ yb,
                            const int* __restrict__ inv,
                            const float* __restrict__ rw,
                            float* __restrict__ out) {
    long long i = (long long)blockIdx.x * blockDim.x + threadIdx.x;
    const long long n4 = (long long)TT * HH / 4;
    if (i >= n4) return;
    long long t  = i / (HH / 4);
    int       c4 = (int)(i % (HH / 4));
    float4 acc; acc.x = 0.f; acc.y = 0.f; acc.z = 0.f; acc.w = 0.f;
    #pragma unroll
    for (int k = 0; k < KK; ++k) {
        int   p = inv[t * KK + k];
        float w = rw [t * KK + k];
        float4 v = ((const float4*)yb)[(size_t)p * (HH / 4) + c4];
        acc.x += w * v.x; acc.y += w * v.y; acc.z += w * v.z; acc.w += w * v.w;
    }
    ((float4*)out)[i] = acc;
}

// ------------------------------- launcher ----------------------------------

extern "C" void kernel_launch(void* const* d_in, const int* in_sizes, int n_in,
                              void* d_out, int out_size, void* d_ws, size_t ws_size,
                              hipStream_t stream) {
    const float* x     = (const float*)d_in[0];
    const int*   sel   = (const int*)  d_in[1];
    const float* rw    = (const float*)d_in[2];
    const float* w1    = (const float*)d_in[3];
    const float* w2    = (const float*)d_in[4];
    const float* w3    = (const float*)d_in[5];
    const float* w1_is = (const float*)d_in[6];
    const float* w2_is = (const float*)d_in[7];
    const float* w3_is = (const float*)d_in[8];
    const float* w1_ws = (const float*)d_in[9];
    const float* w2_ws = (const float*)d_in[10];
    const float* w3_ws = (const float*)d_in[11];
    const float* w1_al = (const float*)d_in[12];
    const float* w2_al = (const float*)d_in[13];
    const float* w3_al = (const float*)d_in[14];
    float* out = (float*)d_out;

    char* ws = (char*)d_ws;
    int*   counts  = (int*)(ws + 0);
    int*   cursors = (int*)(ws + 64);
    int*   offsets = (int*)(ws + 128);
    int*   rows    = (int*)(ws + 256);
    int*   inv     = (int*)(ws + 256 + (size_t)TK * 4);
    size_t o = 256 + (size_t)TK * 8;                       // 131328, 16B aligned
    _Float16* xb  = (_Float16*)(ws + o); o += (size_t)TT * HH * 2;
    _Float16* w1b = (_Float16*)(ws + o); o += (size_t)EE * II * HH * 2;
    _Float16* w3b = (_Float16*)(ws + o); o += (size_t)EE * II * HH * 2;
    _Float16* w2b = (_Float16*)(ws + o); o += (size_t)EE * HH * II * 2;
    _Float16* hb  = (_Float16*)(ws + o); o += (size_t)TK * II * 2;
    float*    yb  = (float*)   (ws + o); o += (size_t)TK * HH * 4;

    // routing
    moe_zero8 <<<1, 32, 0, stream>>>(counts);
    moe_count <<<TK / 256, 256, 0, stream>>>(sel, counts);
    moe_scan  <<<1, 32, 0, stream>>>(counts, offsets, cursors);
    moe_fill  <<<TK / 256, 256, 0, stream>>>(sel, cursors, rows, inv);

    // f32 -> f16 conversion (bandwidth-bound, ~0.7 GB total)
    moe_cvt<<<8192, 256, 0, stream>>>(x,  xb,  (long long)TT * HH / 4);
    moe_cvt<<<8192, 256, 0, stream>>>(w1, w1b, (long long)EE * II * HH / 4);
    moe_cvt<<<8192, 256, 0, stream>>>(w3, w3b, (long long)EE * II * HH / 4);
    moe_cvt<<<8192, 256, 0, stream>>>(w2, w2b, (long long)EE * HH * II / 4);

    // grouped GEMMs (worst-case grid; tiles past counts[e] exit immediately)
    dim3 g1(II / BN, MAXRT, EE);
    moe_stage1<<<g1, 256, 0, stream>>>(xb, w1b, w3b, counts, offsets, rows,
                                       w1_is, w1_ws, w1_al, w3_is, w3_ws, w3_al, hb);
    dim3 g2(HH / BN, MAXRT, EE);
    moe_stage2<<<g2, 256, 0, stream>>>(hb, w2b, counts, offsets,
                                       w2_is, w2_ws, w2_al, yb);

    // deterministic top-K combine
    moe_combine<<<(TT * HH / 4 + 255) / 256, 256, 0, stream>>>(yb, inv, rw, out);
}
